// unit_gcn_69466801045663
// MI455X (gfx1250) — matrix operand order repaired and checked by reading, not compiled, not run on hardware
//
#include <hip/hip_runtime.h>

typedef __attribute__((ext_vector_type(16))) _Float16 v16h;
typedef __attribute__((ext_vector_type(8)))  float    v8f;

#define NM   256
#define TT   128
#define VV   21
#define CC   64
#define TVN  (TT*VV)          // 2688 nodes per nm-block
#define NTOT (NM*TVN)         // 688128 nodes
#define G6   (6*NM*TT)        // 196608: base-edge dst ids are < 6*num_graphs

union AF { v16h v; unsigned u[8]; };

__device__ __forceinline__ unsigned ldsu(const _Float16* p) {
  return *(const unsigned*)p;   // callers guarantee 4B alignment
}

// Load an even-offset f16 channel pair with a single b32, split in registers.
__device__ __forceinline__ float2 ldh2(const _Float16* p) {
  union { unsigned u; _Float16 h[2]; } v;
  v.u = *(const unsigned*)p;
  return make_float2((float)v.h[0], (float)v.h[1]);
}

// ---------------------------------------------------------------------------
// Kernel 1: xl = feat @ Wl^T + bl ; xr = feat @ Wr^T + br  (one conv)
// feat[n][c] = x[nm][c][t][v], n = nm*2688 + (t*21+v). Block = 128 nodes of
// one nm-block (2688/128 = 21 tiles). 8 waves, each owns a 16-node row tile
// and does 4x v_wmma_f32_16x16x32_f16 (K=64 in two steps, for Wl and Wr).
// Results bounce through LDS and leave as coalesced b128 stores.
// ---------------------------------------------------------------------------
__global__ __launch_bounds__(256) void gemm_xlxr(
    const float* __restrict__ x,
    const float* __restrict__ Wl, const float* __restrict__ bl,
    const float* __restrict__ Wr, const float* __restrict__ br,
    _Float16* __restrict__ xlb, _Float16* __restrict__ xrb)
{
  __shared__ __align__(16) _Float16 sF [128*66];  // feat tile, later D bounce
  __shared__ __align__(16) _Float16 sWl[64*66];   // [co][c]
  __shared__ __align__(16) _Float16 sWr[64*66];

  const int t    = threadIdx.x;
  const int nm   = blockIdx.x / 21;
  const int tile = blockIdx.x % 21;
  const float* xb = x + (size_t)nm*CC*TVN + (size_t)tile*128;

  // Load + transpose + f32->f16 a 64ch x 128node tile (coalesced along tv)
  #pragma unroll 4
  for (int it = 0; it < 32; ++it) {
    int idx = it*256 + t;
    int c = idx >> 7, m = idx & 127;
    sF[m*66 + c] = (_Float16)xb[(size_t)c*TVN + m];
  }
  for (int idx = t; idx < 4096; idx += 256) {
    int co = idx >> 6, c = idx & 63;
    sWl[co*66 + c] = (_Float16)Wl[idx];
    sWr[co*66 + c] = (_Float16)Wr[idx];
  }
  __syncthreads();

  const int lane = t & 31, w = t >> 5;
  const int m0   = w * 16;
  const int lsel = lane >> 4;        // K-half selector per ISA A layout
  const int mr   = lane & 15;        // row within tile / output column

  // A fragments (16-bit 16x32 layout): lanes 0-15 halves = K[k0..k0+7],K[k0+16..k0+23]
  //                                    lanes 16-31      = K[k0+8..],K[k0+24..]
  AF a0, a1;
  {
    const _Float16* fr = sF + (m0 + mr)*66 + lsel*8;
    #pragma unroll
    for (int j = 0; j < 4; ++j) { a0.u[j] = ldsu(fr + 2*j); a0.u[4+j] = ldsu(fr + 16 + 2*j); }
    fr += 32;
    #pragma unroll
    for (int j = 0; j < 4; ++j) { a1.u[j] = ldsu(fr + 2*j); a1.u[4+j] = ldsu(fr + 16 + 2*j); }
  }
  // B fragments (16-bit 32x16): lane n = lane&15 holds col n; lanes 0-15 K[k0..k0+7],K[k0+16..k0+23],
  // lanes 16-31 K[k0+8..],K[k0+24..]; B[k][n] = W[n][k] (contiguous in sW row n)
  const int koff = lsel * 8;
  AF fl0, fl1, fr0, fr1;
  {
    const _Float16* p = sWl + mr*66 + koff;
    #pragma unroll
    for (int j = 0; j < 8; ++j) { fl0.u[j] = ldsu(p + 2*j + (j>>2)*8); fl1.u[j] = ldsu(p + 32 + 2*j + (j>>2)*8); }
    const _Float16* q = sWr + mr*66 + koff;
    #pragma unroll
    for (int j = 0; j < 8; ++j) { fr0.u[j] = ldsu(q + 2*j + (j>>2)*8); fr1.u[j] = ldsu(q + 32 + 2*j + (j>>2)*8); }
  }

  v8f cl = {}; v8f cr = {};
  cl = __builtin_amdgcn_wmma_f32_16x16x32_f16(false, a0.v, false, fl0.v, (short)0, cl, false, false);
  cl = __builtin_amdgcn_wmma_f32_16x16x32_f16(false, a1.v, false, fl1.v, (short)0, cl, false, false);
  cr = __builtin_amdgcn_wmma_f32_16x16x32_f16(false, a0.v, false, fr0.v, (short)0, cr, false, false);
  cr = __builtin_amdgcn_wmma_f32_16x16x32_f16(false, a1.v, false, fr1.v, (short)0, cr, false, false);

  const float blv = bl[mr], brv = br[mr];
  const size_t outBase = ((size_t)nm*TVN + (size_t)tile*128) * 64;  // halves

  // C/D layout: VGPR r = row (r + 8*lsel), col = mr.
  // Bounce through LDS (reuse sF, packed [m][co] stride 64) -> b128 stores.
  __syncthreads();                       // all fragment reads of sF done
  #pragma unroll
  for (int r = 0; r < 8; ++r)
    sF[(m0 + lsel*8 + r)*64 + mr] = (_Float16)(cl[r] + blv);
  __syncthreads();
  {
    const uint4* sv = (const uint4*)sF;  // 128*64 halves = 1024 uint4
    uint4* gp = (uint4*)(xlb + outBase);
    #pragma unroll
    for (int j = 0; j < 4; ++j) gp[j*256 + t] = sv[j*256 + t];
  }
  __syncthreads();
  #pragma unroll
  for (int r = 0; r < 8; ++r)
    sF[(m0 + lsel*8 + r)*64 + mr] = (_Float16)(cr[r] + brv);
  __syncthreads();
  {
    const uint4* sv = (const uint4*)sF;
    uint4* gp = (uint4*)(xrb + outBase);
    #pragma unroll
    for (int j = 0; j < 4; ++j) gp[j*256 + t] = sv[j*256 + t];
  }
}

// ---------------------------------------------------------------------------
// Kernel 2: per-dst-node GATv2 attention + aggregation. One wave32 per node,
// 2 channels per lane. dst array is sorted (reference construction), in-degree
// <= 6, so: binary search range, pass1 max, pass2 exp/sum/aggregate.
// ---------------------------------------------------------------------------
__device__ __forceinline__ float rsum32(float v) {
  #pragma unroll
  for (int off = 16; off; off >>= 1) v += __shfl_xor(v, off, 32);
  return v;
}

__global__ __launch_bounds__(256) void gat_edge(
    const _Float16* __restrict__ xlb, const _Float16* __restrict__ xrb,
    const int* __restrict__ srcA, const int* __restrict__ dstA, int E0,
    const float* __restrict__ att, const float* __restrict__ bias,
    float* __restrict__ y, int first)
{
  const int d = (blockIdx.x * blockDim.x + threadIdx.x) >> 5;
  if (d >= NTOT) return;
  const int lane = threadIdx.x & 31;
  const int c = lane * 2;

  const float ax = att[c], ay = att[c+1];
  const float2 xr2  = ldh2(xrb + (size_t)d*64 + c);
  const float2 xld2 = ldh2(xlb + (size_t)d*64 + c);

  // self-loop score
  float h0 = xld2.x + xr2.x; h0 = h0 > 0.f ? h0 : 0.2f*h0;
  float h1 = xld2.y + xr2.y; h1 = h1 > 0.f ? h1 : 0.2f*h1;
  const float e_self = rsum32(ax*h0 + ay*h1);

  int lo = E0;
  if (d < G6) {                       // only these ids have base in-edges
    int a = 0, b = E0;
    while (a < b) { int mid = (a + b) >> 1; if (dstA[mid] < d) a = mid + 1; else b = mid; }
    lo = a;
  }

  // pass 1: running max
  float m = e_self;
  for (int j = lo; j < E0; ++j) {
    if (dstA[j] != d) break;
    int s = srcA[j]; if (s == d) continue;     // removed self-loops masked out
    float2 xs = ldh2(xlb + (size_t)s*64 + c);
    float q0 = xs.x + xr2.x; q0 = q0 > 0.f ? q0 : 0.2f*q0;
    float q1 = xs.y + xr2.y; q1 = q1 > 0.f ? q1 : 0.2f*q1;
    m = fmaxf(m, rsum32(ax*q0 + ay*q1));
  }
  // pass 2: softmax weights + weighted message sum
  float aw   = __expf(e_self - m);
  float wsum = aw;
  float acc0 = aw * xld2.x, acc1 = aw * xld2.y;
  for (int j = lo; j < E0; ++j) {
    if (dstA[j] != d) break;
    int s = srcA[j]; if (s == d) continue;
    float2 xs = ldh2(xlb + (size_t)s*64 + c);
    float q0 = xs.x + xr2.x; q0 = q0 > 0.f ? q0 : 0.2f*q0;
    float q1 = xs.y + xr2.y; q1 = q1 > 0.f ? q1 : 0.2f*q1;
    float e = rsum32(ax*q0 + ay*q1);
    float a2 = __expf(e - m);
    wsum += a2; acc0 += a2*xs.x; acc1 += a2*xs.y;
  }
  const float inv = 1.f / wsum;
  const float o0 = acc0*inv + bias[c];
  const float o1 = acc1*inv + bias[c+1];
  float* py = y + (size_t)d*64 + c;
  if (first) { py[0] = o0;  py[1] = o1;  }
  else       { py[0] += o0; py[1] += o1; }
}

// ---------------------------------------------------------------------------
// Kernel 3: y (node-major) -> BN(gamma=1e-6) + residual x + ReLU -> out in
// [NM,C,T,V] layout. LDS transpose (pad 65) keeps both sides coalesced.
// ---------------------------------------------------------------------------
__global__ __launch_bounds__(256) void bn_res_relu(
    const float* __restrict__ x, const float* __restrict__ y,
    const float* __restrict__ gamma, const float* __restrict__ beta,
    float* __restrict__ out)
{
  __shared__ float sY[32*65];
  const int t   = threadIdx.x;
  const int nm  = blockIdx.x / 84;       // TVN/32 = 84 tiles per nm
  const int tb  = blockIdx.x % 84;
  const int tv0 = tb * 32;
  const float* yb = y + ((size_t)nm*TVN + tv0) * 64;

  #pragma unroll
  for (int it = 0; it < 8; ++it) {       // 32 nodes x 64 ch, fully coalesced
    int idx = it*256 + t;
    sY[(idx >> 6)*65 + (idx & 63)] = yb[idx];
  }
  __syncthreads();

  const float invs = rsqrtf(1.f + 1e-5f);
  #pragma unroll
  for (int it = 0; it < 8; ++it) {
    int idx = it*256 + t;
    int ch = idx >> 5, j = idx & 31;     // consecutive t -> consecutive tv
    size_t gx = ((size_t)nm*64 + ch)*TVN + tv0 + j;
    float v = sY[j*65 + ch] * invs * gamma[ch] + beta[ch] + x[gx];
    out[gx] = v > 0.f ? v : 0.f;
  }
}

// ---------------------------------------------------------------------------
extern "C" void kernel_launch(void* const* d_in, const int* in_sizes, int n_in,
                              void* d_out, int out_size, void* d_ws, size_t ws_size,
                              hipStream_t stream)
{
  const float* x     = (const float*)d_in[0];
  const int*   src   = (const int*)  d_in[1];
  const int*   dst   = (const int*)  d_in[2];
  const float* Wl    = (const float*)d_in[3];
  const float* bl    = (const float*)d_in[4];
  const float* Wr    = (const float*)d_in[5];
  const float* br    = (const float*)d_in[6];
  const float* att   = (const float*)d_in[7];
  const float* bias  = (const float*)d_in[8];
  const float* gamma = (const float*)d_in[9];
  const float* beta  = (const float*)d_in[10];
  const int E0 = in_sizes[1];

  // workspace: xl (f16, 88MB) | xr (f16, 88MB) | y (f32, 176MB)
  _Float16* xlb = (_Float16*)d_ws;
  _Float16* xrb = xlb + (size_t)NTOT * 64;
  float*    y   = (float*)((char*)d_ws + (size_t)2 * NTOT * 64 * sizeof(_Float16));

  for (int i = 0; i < 3; ++i) {
    gemm_xlxr<<<NM*21, 256, 0, stream>>>(x, Wl + i*4096, bl + i*64,
                                         Wr + i*4096, br + i*64, xlb, xrb);
    gat_edge<<<NTOT/8, 256, 0, stream>>>(xlb, xrb, src, dst, E0,
                                         att + i*64, bias + i*64, y, i == 0);
  }
  bn_res_relu<<<NM*84, 256, 0, stream>>>(x, y, gamma, beta, (float*)d_out);
}
